// MLAAttention_48928267436394
// MI455X (gfx1250) — matrix operand order repaired and checked by reading, not compiled, not run on hardware
//
#include <hip/hip_runtime.h>

#define S_LEN    2048
#define BATCH    4
#define D_MODELC 2048
#define NUM_H    16
#define NUM_KV   4
#define HEAD_D   128
#define D_LAT    512
#define MROWS    (S_LEN * BATCH)   // 8192

typedef __attribute__((ext_vector_type(16))) __bf16 v16bf;
typedef __attribute__((ext_vector_type(8)))  float  v8f;

union Frag { v16bf v; unsigned int u[8]; };

__device__ inline unsigned short f32_to_bf16(float f) {
  unsigned int u = __float_as_uint(f);
  u += 0x7FFFu + ((u >> 16) & 1u);   // round-to-nearest-even
  return (unsigned short)(u >> 16);
}
__device__ inline float bf16_to_f32(unsigned short h) {
  return __uint_as_float(((unsigned int)h) << 16);
}
__device__ inline v8f vzero8() {
  v8f z;
#pragma unroll
  for (int i = 0; i < 8; ++i) z[i] = 0.f;
  return z;
}
__device__ inline v8f wmma_bf16(const Frag& a, const Frag& b, v8f c) {
  return __builtin_amdgcn_wmma_f32_16x16x32_bf16(
      /*neg_a=*/false, a.v, /*neg_b=*/false, b.v,
      /*c_mod=*/(short)0, c, /*reuse_a=*/false, /*reuse_b=*/false);
}

// Async VMEM -> LDS copy, 16 bytes per lane, tracked by ASYNCcnt (CDNA5).
// lds_lo32: low 32 bits of the flat LDS address.
__device__ inline void async_copy_b128(unsigned lds_lo32, const void* gaddr) {
  asm volatile("global_load_async_to_lds_b128 %0, %1, off"
               :: "v"(lds_lo32), "v"((unsigned long long)(size_t)gaddr)
               : "memory");
}
__device__ inline void wait_asynccnt0() {
  asm volatile("s_wait_asynccnt 0" ::: "memory");
}

// ---------------------------------------------------------------- f32 -> bf16
__global__ __launch_bounds__(256) void cvt_f32_bf16(const float* __restrict__ in,
                                                    unsigned short* __restrict__ out,
                                                    int n) {
  int i = blockIdx.x * 256 + threadIdx.x;
  if (i < n) out[i] = f32_to_bf16(in[i]);
}

// ---------------------------------------------------------------- tiled GEMM
// C(M,N) = A(M,K) * B(K,N) + bias ; A,B bf16 row-major ; out f32 or bf16.
// 64x64 block tile, 4 waves (32x32 each), BK=32 (one WMMA per K-slab).
// A tile staged with GLOBAL_LOAD_ASYNC_TO_LDS_B128 (no VGPR round trip);
// B tile goes through VGPRs because it is transposed into [n][k] layout.
template <bool OUT_BF16>
__global__ __launch_bounds__(128) void gemm_bf16(const unsigned short* __restrict__ A,
                                                 const unsigned short* __restrict__ Bm,
                                                 const float* __restrict__ bias,
                                                 void* __restrict__ Cout,
                                                 int Mdim, int Ndim, int Kdim) {
  __shared__ __align__(16) unsigned short sA[64 * 32];   // [m][k]
  __shared__ __align__(16) unsigned short sB[64 * 32];   // transposed: [n][k]
  const int t    = threadIdx.x;
  const int wid  = t >> 5;
  const int lane = t & 31;
  const int lm   = lane & 15;
  const int hi   = lane >> 4;
  const int m0   = blockIdx.y * 64;
  const int n0   = blockIdx.x * 64;
  const int wr   = wid >> 1;          // 0..1
  const int wc   = wid & 1;           // 0..1

  v8f acc[2][2];
#pragma unroll
  for (int i = 0; i < 2; ++i)
#pragma unroll
    for (int j = 0; j < 2; ++j) acc[i][j] = vzero8();

  for (int kb = 0; kb < Kdim; kb += 32) {
    // A tile 64x32: 256 16-byte chunks, 2 per thread, async straight into LDS
#pragma unroll
    for (int i = 0; i < 2; ++i) {
      int c = t + 128 * i;
      int r = c >> 2, col = (c & 3) * 8;
      const unsigned short* g = A + (size_t)(m0 + r) * Kdim + kb + col;
      async_copy_b128((unsigned)(size_t)(sA + r * 32 + col), g);
    }
    // B tile 32x64, stored transposed [n][k] so frag K-pairs are contiguous
#pragma unroll
    for (int i = 0; i < 2; ++i) {
      int c = t + 128 * i;
      int r = c >> 3, col = (c & 7) * 8;   // r = k row, col = n
      uint4 vv = *(const uint4*)(Bm + (size_t)(kb + r) * Ndim + n0 + col);
      const unsigned short* pv = (const unsigned short*)&vv;
#pragma unroll
      for (int j = 0; j < 8; ++j) sB[(col + j) * 32 + r] = pv[j];
    }
    if (kb + 32 < Kdim) {   // hint next slabs toward L2 (global_prefetch_b8)
      __builtin_prefetch(A + (size_t)(m0 + (t >> 1)) * Kdim + kb + 32, 0, 1);
      __builtin_prefetch(Bm + (size_t)(kb + 32 + (t & 31)) * Ndim + n0, 0, 1);
    }
    wait_asynccnt0();        // async A-tile landed in LDS
    __syncthreads();

    Frag af[2], bf[2];
#pragma unroll
    for (int tr = 0; tr < 2; ++tr) {
      int m = wr * 32 + tr * 16 + lm;
#pragma unroll
      for (int p = 0; p < 8; ++p) {
        int k = 16 * hi + 2 * p;
        af[tr].u[p] = *(const unsigned int*)(sA + m * 32 + k);
      }
    }
#pragma unroll
    for (int tc = 0; tc < 2; ++tc) {
      int n = wc * 32 + tc * 16 + lm;
#pragma unroll
      for (int p = 0; p < 8; ++p) {
        bf[tc].u[p] = *(const unsigned int*)(sB + n * 32 + 16 * hi + 2 * p);
      }
    }
#pragma unroll
    for (int tr = 0; tr < 2; ++tr)
#pragma unroll
      for (int tc = 0; tc < 2; ++tc)
        acc[tr][tc] = wmma_bf16(af[tr], bf[tc], acc[tr][tc]);
    __syncthreads();
  }

  // epilogue: C layout lane = N column, VGPR r -> M = r + 8*hi
#pragma unroll
  for (int tr = 0; tr < 2; ++tr)
#pragma unroll
    for (int tc = 0; tc < 2; ++tc) {
      int gn = n0 + wc * 32 + tc * 16 + lm;
      float bv = bias[gn];
#pragma unroll
      for (int r = 0; r < 8; ++r) {
        int gm = m0 + wr * 32 + tr * 16 + 8 * hi + r;
        float val = acc[tr][tc][r] + bv;
        if (OUT_BF16)
          ((unsigned short*)Cout)[(size_t)gm * Ndim + gn] = f32_to_bf16(val);
        else
          ((float*)Cout)[(size_t)gm * Ndim + gn] = val;
      }
    }
}

// ---------------------------------------------------------------- RoPE + reorder
// qin (S,B,H*128) bf16 -> qout (B*H, S, 128) bf16, rotated
__global__ __launch_bounds__(64) void rope_q(const unsigned short* __restrict__ qin,
                                             unsigned short* __restrict__ qout) {
  int s = blockIdx.x, bh = blockIdx.y, d = threadIdx.x;  // d in [0,64)
  int b = bh >> 4, h = bh & 15;
  const unsigned short* p = qin + ((size_t)s * BATCH + b) * D_MODELC + h * HEAD_D;
  float t1 = bf16_to_f32(p[d]);
  float t2 = bf16_to_f32(p[d + 64]);
  float inv = __expf(-(float)(2 * d) * (1.0f / 128.f) * 9.210340371976184f); // ln(1e4)
  float ang = (float)s * inv, c, sn;
  __sincosf(ang, &sn, &c);
  unsigned short* q = qout + ((size_t)bh * S_LEN + s) * HEAD_D;
  q[d]      = f32_to_bf16(t1 * c - t2 * sn);
  q[d + 64] = f32_to_bf16(t1 * sn + t2 * c);
}

// kin (S,B,4*128) bf16 -> kout (B*4, S, 128) bf16, rotated
__global__ __launch_bounds__(64) void rope_k(const unsigned short* __restrict__ kin,
                                             unsigned short* __restrict__ kout) {
  int s = blockIdx.x, bh = blockIdx.y, d = threadIdx.x;
  int b = bh >> 2, h = bh & 3;
  const unsigned short* p = kin + ((size_t)s * BATCH + b) * D_LAT + h * HEAD_D;
  float t1 = bf16_to_f32(p[d]);
  float t2 = bf16_to_f32(p[d + 64]);
  float inv = __expf(-(float)(2 * d) * (1.0f / 128.f) * 9.210340371976184f);
  float ang = (float)s * inv, c, sn;
  __sincosf(ang, &sn, &c);
  unsigned short* q = kout + ((size_t)bh * S_LEN + s) * HEAD_D;
  q[d]      = f32_to_bf16(t1 * c - t2 * sn);
  q[d + 64] = f32_to_bf16(t1 * sn + t2 * c);
}

// vin (S,B,512) bf16 -> vout (B*4, 128, S) bf16  (V transposed: d-major)
__global__ __launch_bounds__(128) void vtrans(const unsigned short* __restrict__ vin,
                                              unsigned short* __restrict__ vout) {
  int s = blockIdx.x, bh = blockIdx.y, d = threadIdx.x;
  int b = bh >> 2, h = bh & 3;
  vout[((size_t)bh * HEAD_D + d) * S_LEN + s] =
      vin[((size_t)s * BATCH + b) * D_LAT + h * HEAD_D + d];
}

// ---------------------------------------------------------------- flash attention
// q_r (B*H,S,128) ; k_r (B*4,S,128) ; v_t (B*4,128,S) ; ao (S,B,2048) all bf16
__global__ __launch_bounds__(128) void flash_attn(const unsigned short* __restrict__ q_r,
                                                  const unsigned short* __restrict__ k_r,
                                                  const unsigned short* __restrict__ v_t,
                                                  unsigned short* __restrict__ ao) {
  __shared__ __align__(16) unsigned short sP[4][16 * 32];   // per-wave P tile
  const int wid  = threadIdx.x >> 5;
  const int lane = threadIdx.x & 31;
  const int lm   = lane & 15;
  const int hi   = lane >> 4;
  const int bh   = blockIdx.y;          // 0..63
  const int b    = bh >> 4;
  const int h    = bh & 15;
  const int hk   = h >> 2;              // GQA group of 4
  const int q0   = blockIdx.x * 64 + wid * 16;

  const unsigned short* qp = q_r + (size_t)bh * S_LEN * HEAD_D;
  const unsigned short* kp = k_r + (size_t)(b * NUM_KV + hk) * S_LEN * HEAD_D;
  const unsigned short* vp = v_t + (size_t)(b * NUM_KV + hk) * HEAD_D * S_LEN;

  // Q fragments: 4 chunks of K=32 over head_dim, kept resident
  Frag qf[4];
#pragma unroll
  for (int dc = 0; dc < 4; ++dc)
#pragma unroll
    for (int p = 0; p < 8; ++p) {
      int k = 32 * dc + 16 * hi + 2 * p;
      qf[dc].u[p] = *(const unsigned int*)(qp + (size_t)(q0 + lm) * HEAD_D + k);
    }

  v8f o[8];
  float mrow[8], lrow[8];
#pragma unroll
  for (int t = 0; t < 8; ++t) o[t] = vzero8();
#pragma unroll
  for (int r = 0; r < 8; ++r) { mrow[r] = -1e30f; lrow[r] = 0.f; }

  const float scale = 0.08838834764831845f;   // 1/sqrt(128)
  const int jmax = (q0 + 15) / 32;            // causal extent (inclusive)

  for (int j = 0; j <= jmax; ++j) {
    const int kb = j * 32;
    v8f s0 = vzero8(), s1 = vzero8();
#pragma unroll
    for (int dc = 0; dc < 4; ++dc) {
      Frag b0, b1;
#pragma unroll
      for (int p = 0; p < 8; ++p) {
        int d = 32 * dc + 16 * hi + 2 * p;
        b0.u[p] = *(const unsigned int*)(kp + (size_t)(kb + lm) * HEAD_D + d);
        b1.u[p] = *(const unsigned int*)(kp + (size_t)(kb + 16 + lm) * HEAD_D + d);
      }
      s0 = wmma_bf16(qf[dc], b0, s0);
      s1 = wmma_bf16(qf[dc], b1, s1);
    }

    // online softmax (rows r+8*hi ; lane = key column within half)
#pragma unroll
    for (int r = 0; r < 8; ++r) {
      int qi = q0 + 8 * hi + r;
      int k0 = kb + lm, k1 = k0 + 16;
      float a0 = (k0 <= qi) ? s0[r] * scale : -1e30f;
      float a1 = (k1 <= qi) ? s1[r] * scale : -1e30f;
      float mx = fmaxf(a0, a1);
      mx = fmaxf(mx, __shfl_xor(mx, 1));
      mx = fmaxf(mx, __shfl_xor(mx, 2));
      mx = fmaxf(mx, __shfl_xor(mx, 4));
      mx = fmaxf(mx, __shfl_xor(mx, 8));
      float mnew  = fmaxf(mrow[r], mx);
      float alpha = __expf(mrow[r] - mnew);
      float p0 = __expf(a0 - mnew);
      float p1 = __expf(a1 - mnew);
      float rs = p0 + p1;
      rs += __shfl_xor(rs, 1);
      rs += __shfl_xor(rs, 2);
      rs += __shfl_xor(rs, 4);
      rs += __shfl_xor(rs, 8);
      lrow[r] = lrow[r] * alpha + rs;
      mrow[r] = mnew;
#pragma unroll
      for (int t = 0; t < 8; ++t) o[t][r] = o[t][r] * alpha;
      sP[wid][(8 * hi + r) * 32 + lm]      = f32_to_bf16(p0);
      sP[wid][(8 * hi + r) * 32 + 16 + lm] = f32_to_bf16(p1);
    }

    asm volatile("s_wait_dscnt 0" ::: "memory");   // wave-private LDS RAW

    // P as one 16x32 A-fragment
    Frag pf;
#pragma unroll
    for (int p = 0; p < 8; ++p) {
      int k = 16 * hi + 2 * p;
      pf.u[p] = *(const unsigned int*)(&sP[wid][lm * 32 + k]);
    }
    // P(16x32) x V(32x16) per 16-wide output d-tile; V transposed => contiguous pairs
#pragma unroll
    for (int t = 0; t < 8; ++t) {
      Frag vf;
#pragma unroll
      for (int p = 0; p < 8; ++p)
        vf.u[p] = *(const unsigned int*)(vp + (size_t)(16 * t + lm) * S_LEN + kb + 16 * hi + 2 * p);
      o[t] = wmma_bf16(pf, vf, o[t]);
    }
  }

  // normalize and scatter into (S,B,2048) bf16 for the output projection
#pragma unroll
  for (int r = 0; r < 8; ++r) {
    float inv = 1.f / lrow[r];
    int qi = q0 + 8 * hi + r;
#pragma unroll
    for (int t = 0; t < 8; ++t)
      ao[((size_t)qi * BATCH + b) * D_MODELC + h * HEAD_D + 16 * t + lm] =
          f32_to_bf16(o[t][r] * inv);
  }
}

// ---------------------------------------------------------------- launcher
extern "C" void kernel_launch(void* const* d_in, const int* in_sizes, int n_in,
                              void* d_out, int out_size, void* d_ws, size_t ws_size,
                              hipStream_t stream) {
  const float* x  = (const float*)d_in[0];
  // d_in[1] = causal_mask (computed analytically in-kernel)
  const float* Wq = (const float*)d_in[2];
  const float* bq = (const float*)d_in[3];
  const float* Wc = (const float*)d_in[4];
  const float* bc = (const float*)d_in[5];
  const float* Wk = (const float*)d_in[6];
  const float* bk = (const float*)d_in[7];
  const float* Wv = (const float*)d_in[8];
  const float* bv = (const float*)d_in[9];
  const float* Wo = (const float*)d_in[10];
  const float* bo = (const float*)d_in[11];

  char* ws = (char*)d_ws;
  size_t off = 0;
  auto carve = [&](size_t bytes) {
    char* p = ws + off;
    off += (bytes + 255) & ~(size_t)255;
    return p;
  };
  unsigned short* xb   = (unsigned short*)carve((size_t)MROWS * D_MODELC * 2);
  unsigned short* wq_b = (unsigned short*)carve((size_t)D_MODELC * D_MODELC * 2);
  unsigned short* wc_b = (unsigned short*)carve((size_t)D_MODELC * D_LAT * 2);
  unsigned short* wk_b = (unsigned short*)carve((size_t)D_LAT * D_LAT * 2);
  unsigned short* wv_b = (unsigned short*)carve((size_t)D_LAT * D_LAT * 2);
  unsigned short* wo_b = (unsigned short*)carve((size_t)D_MODELC * D_MODELC * 2);
  unsigned short* q_b  = (unsigned short*)carve((size_t)MROWS * D_MODELC * 2);
  unsigned short* lat  = (unsigned short*)carve((size_t)MROWS * D_LAT * 2);
  unsigned short* k_b  = (unsigned short*)carve((size_t)MROWS * D_LAT * 2);
  unsigned short* v_b  = (unsigned short*)carve((size_t)MROWS * D_LAT * 2);
  unsigned short* q_r  = (unsigned short*)carve((size_t)MROWS * D_MODELC * 2);
  unsigned short* k_r  = (unsigned short*)carve((size_t)BATCH * NUM_KV * S_LEN * HEAD_D * 2);
  unsigned short* v_t  = (unsigned short*)carve((size_t)BATCH * NUM_KV * HEAD_D * S_LEN * 2);
  unsigned short* ao   = (unsigned short*)carve((size_t)MROWS * D_MODELC * 2);

  auto cvt = [&](const float* in, unsigned short* out, int n) {
    cvt_f32_bf16<<<(n + 255) / 256, 256, 0, stream>>>(in, out, n);
  };
  cvt(x,  xb,   MROWS * D_MODELC);
  cvt(Wq, wq_b, D_MODELC * D_MODELC);
  cvt(Wc, wc_b, D_MODELC * D_LAT);
  cvt(Wk, wk_b, D_LAT * D_LAT);
  cvt(Wv, wv_b, D_LAT * D_LAT);
  cvt(Wo, wo_b, D_MODELC * D_MODELC);

  // Q / latent / K / V projections (bf16 out)
  gemm_bf16<true><<<dim3(D_MODELC / 64, MROWS / 64), 128, 0, stream>>>(
      xb, wq_b, bq, (void*)q_b, MROWS, D_MODELC, D_MODELC);
  gemm_bf16<true><<<dim3(D_LAT / 64, MROWS / 64), 128, 0, stream>>>(
      xb, wc_b, bc, (void*)lat, MROWS, D_LAT, D_MODELC);
  gemm_bf16<true><<<dim3(D_LAT / 64, MROWS / 64), 128, 0, stream>>>(
      lat, wk_b, bk, (void*)k_b, MROWS, D_LAT, D_LAT);
  gemm_bf16<true><<<dim3(D_LAT / 64, MROWS / 64), 128, 0, stream>>>(
      lat, wv_b, bv, (void*)v_b, MROWS, D_LAT, D_LAT);

  // RoPE + layout transforms
  rope_q<<<dim3(S_LEN, BATCH * NUM_H), 64, 0, stream>>>(q_b, q_r);
  rope_k<<<dim3(S_LEN, BATCH * NUM_KV), 64, 0, stream>>>(k_b, k_r);
  vtrans<<<dim3(S_LEN, BATCH * NUM_KV), 128, 0, stream>>>(v_b, v_t);

  // causal flash attention
  flash_attn<<<dim3(S_LEN / 64, BATCH * NUM_H), 128, 0, stream>>>(q_r, k_r, v_t, ao);

  // output projection (f32 out -> d_out)
  gemm_bf16<false><<<dim3(D_MODELC / 64, MROWS / 64), 128, 0, stream>>>(
      ao, wo_b, bo, d_out, MROWS, D_MODELC, D_MODELC);
}
